// GRIC_31550829756424
// MI455X (gfx1250) — matrix operand order repaired and checked by reading, not compiled, not run on hardware
//
#include <hip/hip_runtime.h>
#include <hip/hip_bf16.h>
#include <stdint.h>

// ---------------------------------------------------------------------------
// CDNA5 (gfx1250) wave32 WMMA implementation of the graph-transformer layer.
// All GEMMs run as C = A * B^T on bf16 fragments with f32 accumulation via
// v_wmma_f32_16x16x32_bf16.  The big GEMMs stage the block-shared A tile in
// LDS with async global->LDS copies (ASYNCcnt / s_wait_asynccnt path).
// LN / softmax / epilogues are f32.
// ---------------------------------------------------------------------------

typedef __attribute__((ext_vector_type(16))) __bf16 bf16x16;
typedef __attribute__((ext_vector_type(8)))  float  f32x8;

union FragBF { bf16x16 v; uint4 q[2]; };

__device__ __forceinline__ float xmax16(float v) {
  v = fmaxf(v, __shfl_xor(v, 1, 32));
  v = fmaxf(v, __shfl_xor(v, 2, 32));
  v = fmaxf(v, __shfl_xor(v, 4, 32));
  v = fmaxf(v, __shfl_xor(v, 8, 32));
  return v;
}
__device__ __forceinline__ float xsum16(float v) {
  v += __shfl_xor(v, 1, 32);
  v += __shfl_xor(v, 2, 32);
  v += __shfl_xor(v, 4, 32);
  v += __shfl_xor(v, 8, 32);
  return v;
}
__device__ __forceinline__ float xsum32(float v) {
  v = xsum16(v);
  v += __shfl_xor(v, 16, 32);
  return v;
}

// ---------------------------------------------------------------------------
// Weight transpose + f32 -> bf16 : Wt[f*K + k] = W[k*F + f]
// ---------------------------------------------------------------------------
__global__ __launch_bounds__(256)
void k_w_transpose(const float* __restrict__ W, __bf16* __restrict__ Wt,
                   int K, int F) {
  int i = blockIdx.x * 256 + threadIdx.x;
  if (i >= K * F) return;
  int f = i / K;
  int k = i - f * K;
  Wt[i] = (__bf16)W[(size_t)k * F + f];
}

// ---------------------------------------------------------------------------
// Per-row (one wave / row):  Ahat = A with diagonal := 1 (bf16 out),
// dis = rsqrt(max(rowsum(Ahat), 1))
// ---------------------------------------------------------------------------
__global__ __launch_bounds__(256)
void k_adj_prep(const float* __restrict__ A, __bf16* __restrict__ Abf,
                float* __restrict__ dis) {
  const int N = 1024;
  const int lane = threadIdx.x & 31;
  const int rowg = blockIdx.x * 8 + (threadIdx.x >> 5); // b*N + n
  const int n = rowg & (N - 1);
  const float* ar = A + (size_t)rowg * N;
  const int base = lane * 32;
  const float4* p4 = (const float4*)(ar + base);
  float vals[32];
  float s = 0.0f;
#pragma unroll
  for (int j = 0; j < 8; ++j) {
    float4 t = p4[j];
    int c = base + 4 * j;
    if (c + 0 == n) t.x = 1.0f;
    if (c + 1 == n) t.y = 1.0f;
    if (c + 2 == n) t.z = 1.0f;
    if (c + 3 == n) t.w = 1.0f;
    vals[4 * j + 0] = t.x; vals[4 * j + 1] = t.y;
    vals[4 * j + 2] = t.z; vals[4 * j + 3] = t.w;
    s += t.x + t.y + t.z + t.w;
  }
  s = xsum32(s);
  if (lane == 0) dis[rowg] = rsqrtf(fmaxf(s, 1.0f));
  __bf16 pk[32];
#pragma unroll
  for (int j = 0; j < 32; ++j) pk[j] = (__bf16)vals[j];
  uint4* q = (uint4*)(Abf + (size_t)rowg * N + base);
#pragma unroll
  for (int j = 0; j < 4; ++j) q[j] = ((const uint4*)pk)[j];
}

// ---------------------------------------------------------------------------
// LayerNorm over 128 features, one wave per row.
// ---------------------------------------------------------------------------
template<bool WF32, bool WBF16>
__global__ __launch_bounds__(256)
void k_ln128(const float* __restrict__ X, const float* __restrict__ g,
             const float* __restrict__ be,
             float* __restrict__ Of, __bf16* __restrict__ Ob, int rows) {
  const int lane = threadIdx.x & 31;
  const int row = blockIdx.x * 8 + (threadIdx.x >> 5);
  if (row >= rows) return;
  const float4 v = *(const float4*)(X + (size_t)row * 128 + lane * 4);
  float s  = v.x + v.y + v.z + v.w;
  float s2 = v.x * v.x + v.y * v.y + v.z * v.z + v.w * v.w;
  s = xsum32(s); s2 = xsum32(s2);
  const float mean = s * (1.0f / 128.0f);
  const float var  = s2 * (1.0f / 128.0f) - mean * mean;
  const float rstd = rsqrtf(var + 1e-5f);
  const float4 gg = *(const float4*)(g + lane * 4);
  const float4 bb = *(const float4*)(be + lane * 4);
  float o0 = (v.x - mean) * rstd * gg.x + bb.x;
  float o1 = (v.y - mean) * rstd * gg.y + bb.y;
  float o2 = (v.z - mean) * rstd * gg.z + bb.z;
  float o3 = (v.w - mean) * rstd * gg.w + bb.w;
  if (WF32)
    *(float4*)(Of + (size_t)row * 128 + lane * 4) = make_float4(o0, o1, o2, o3);
  if (WBF16) {
    __bf16 pk[4] = { (__bf16)o0, (__bf16)o1, (__bf16)o2, (__bf16)o3 };
    *(uint2*)(Ob + (size_t)row * 128 + lane * 4) = *(const uint2*)pk;
  }
}

// out = Oin + LN(X)
__global__ __launch_bounds__(256)
void k_ln_add(const float* __restrict__ X, const float* __restrict__ g,
              const float* __restrict__ be, const float* __restrict__ Oin,
              float* __restrict__ out, int rows) {
  const int lane = threadIdx.x & 31;
  const int row = blockIdx.x * 8 + (threadIdx.x >> 5);
  if (row >= rows) return;
  const float4 v = *(const float4*)(X + (size_t)row * 128 + lane * 4);
  float s  = v.x + v.y + v.z + v.w;
  float s2 = v.x * v.x + v.y * v.y + v.z * v.z + v.w * v.w;
  s = xsum32(s); s2 = xsum32(s2);
  const float mean = s * (1.0f / 128.0f);
  const float var  = s2 * (1.0f / 128.0f) - mean * mean;
  const float rstd = rsqrtf(var + 1e-5f);
  const float4 gg = *(const float4*)(g + lane * 4);
  const float4 bb = *(const float4*)(be + lane * 4);
  const float4 ov = *(const float4*)(Oin + (size_t)row * 128 + lane * 4);
  float4 r;
  r.x = ov.x + (v.x - mean) * rstd * gg.x + bb.x;
  r.y = ov.y + (v.y - mean) * rstd * gg.y + bb.y;
  r.z = ov.z + (v.z - mean) * rstd * gg.z + bb.z;
  r.w = ov.w + (v.w - mean) * rstd * gg.w + bb.w;
  *(float4*)(out + (size_t)row * 128 + lane * 4) = r;
}

// ---------------------------------------------------------------------------
// Shared GEMM epilogue.  C tile layout (per CDNA5 ISA):
//   vgpr r, lane l -> row rowb + r  (rowb = tM*16 + 8*(l>>4)), col tNg*64+nt*16+(l&15)
// ---------------------------------------------------------------------------
enum { EP_SCALE_T = 0, EP_SCALE_BIAS = 1, EP_SCALE_BIAS_T = 2,
       EP_RESID_F32 = 3, EP_BIAS_RELU_BF = 4, EP_BIAS_RELU_F32 = 5 };

template<int EP>
__device__ __forceinline__ void gemm_epilogue(
    f32x8 (&acc)[4], void* __restrict__ Cp, long long sC, int ldc, int b,
    const float* __restrict__ dis, const float* __restrict__ bias,
    const float* __restrict__ resid, int ldr, int rowb, int tNg, int lq) {
#pragma unroll
  for (int nt = 0; nt < 4; ++nt) {
    const int col = tNg * 64 + nt * 16 + lq;
    if constexpr (EP == EP_SCALE_T || EP == EP_SCALE_BIAS_T) {
      // transposed, per-lane contiguous b128 store: Ct[col, rowb..rowb+7]
      const float bc = (EP == EP_SCALE_BIAS_T) ? bias[col] : 0.0f;
      __bf16 pk[8];
#pragma unroll
      for (int r = 0; r < 8; ++r)
        pk[r] = (__bf16)(acc[nt][r] * dis[rowb + r] + bc);
      __bf16* Ct = (__bf16*)Cp + (size_t)b * sC;
      *(uint4*)(Ct + (size_t)col * ldc + rowb) = *(const uint4*)pk;
    } else if constexpr (EP == EP_SCALE_BIAS) {
      __bf16* C = (__bf16*)Cp + (size_t)b * sC;
      const float bc = bias[col];
#pragma unroll
      for (int r = 0; r < 8; ++r)
        C[(size_t)(rowb + r) * ldc + col] =
            (__bf16)(acc[nt][r] * dis[rowb + r] + bc);
    } else if constexpr (EP == EP_RESID_F32) {
      float* C = (float*)Cp + (size_t)b * sC;
#pragma unroll
      for (int r = 0; r < 8; ++r)
        C[(size_t)(rowb + r) * ldc + col] =
            acc[nt][r] + resid[(size_t)(rowb + r) * ldr + col];
    } else if constexpr (EP == EP_BIAS_RELU_BF) {
      __bf16* C = (__bf16*)Cp + (size_t)b * sC;
      const float bc = bias[col];
#pragma unroll
      for (int r = 0; r < 8; ++r)
        C[(size_t)(rowb + r) * ldc + col] =
            (__bf16)fmaxf(acc[nt][r] + bc, 0.0f);
    } else {
      float* C = (float*)Cp + (size_t)b * sC;
      const float bc = bias[col];
#pragma unroll
      for (int r = 0; r < 8; ++r)
        C[(size_t)(rowb + r) * ldc + col] = fmaxf(acc[nt][r] + bc, 0.0f);
    }
  }
}

// ---------------------------------------------------------------------------
// Generic WMMA GEMM (direct global fragment loads):
//   C = A[M,K] * Bt[Ncols,K]^T, bf16 row-major operands, f32 accumulate.
// Wave = 16x64 tile (4 accumulators); 8 waves / block.
//   A frag: lane l holds row (l&15), K chunk  k0+16*(l>>4)+[0..15] (contiguous)
//   B frag: lane l holds col (l&15), K chunk  k0+16*(l>>4)+[0..15] (contiguous)
// ---------------------------------------------------------------------------
template<int EP>
__global__ __launch_bounds__(256)
void k_gemm_bt(const __bf16* __restrict__ A, long long sA, int lda,
               const __bf16* __restrict__ Bt, long long sB, int ldb,
               void* __restrict__ Cp, long long sC, int ldc,
               const float* __restrict__ dis, long long sD,
               const float* __restrict__ bias,
               const float* __restrict__ resid, long long sR, int ldr,
               int M, int Ncols, int K) {
  const int lane = threadIdx.x & 31, half = lane >> 4, lq = lane & 15;
  const int b = blockIdx.y;
  A  += (size_t)b * sA;
  Bt += (size_t)b * sB;
  if (dis)   dis   += (size_t)b * sD;
  if (resid) resid += (size_t)b * sR;

  const int ngrp = Ncols >> 6;                    // 64-col groups
  const int wid  = blockIdx.x * 8 + (threadIdx.x >> 5);
  const int tM   = wid / ngrp;
  const int tNg  = wid % ngrp;
  if (tM * 16 >= M) return;                       // wave-uniform

  f32x8 acc[4] = {};
  const __bf16* pa = A + (size_t)(tM * 16 + lq) * lda;
  const __bf16* pb[4];
#pragma unroll
  for (int nt = 0; nt < 4; ++nt)
    pb[nt] = Bt + (size_t)(tNg * 64 + nt * 16 + lq) * ldb;

  for (int k0 = 0; k0 < K; k0 += 32) {
    FragBF a;
    a.q[0] = *(const uint4*)(pa + k0 + half * 16);
    a.q[1] = *(const uint4*)(pa + k0 + half * 16 + 8);
#pragma unroll
    for (int nt = 0; nt < 4; ++nt) {
      FragBF bb;
      bb.q[0] = *(const uint4*)(pb[nt] + k0 + half * 16);
      bb.q[1] = *(const uint4*)(pb[nt] + k0 + half * 16 + 8);
      acc[nt] = __builtin_amdgcn_wmma_f32_16x16x32_bf16(
          false, a.v, false, bb.v, (short)0, acc[nt], false, false);
    }
  }
  gemm_epilogue<EP>(acc, Cp, sC, ldc, b, dis, bias, resid, ldr,
                    tM * 16 + half * 8, tNg, lq);
}

// ---------------------------------------------------------------------------
// WMMA GEMM with async-LDS staging of the block-shared A tile.
// Requires ngrp = Ncols/64 to be a multiple of 8 so all 8 waves of a block
// share tM (true for Ncols = 1024).  Wave 0 copies the 16x32 bf16 A tile
// into a double-buffered LDS stage with GLOBAL_LOAD_ASYNC_TO_LDS_B128
// (ASYNCcnt), overlapping the next tile's copy with this tile's WMMAs.
// ---------------------------------------------------------------------------
template<int EP>
__global__ __launch_bounds__(256)
void k_gemm_adj(const __bf16* __restrict__ A, long long sA, int lda,
                const __bf16* __restrict__ Bt, long long sB, int ldb,
                void* __restrict__ Cp, long long sC, int ldc,
                const float* __restrict__ dis, long long sD,
                const float* __restrict__ bias,
                const float* __restrict__ resid, long long sR, int ldr,
                int Ncols, int K) {
  const int lane = threadIdx.x & 31, half = lane >> 4, lq = lane & 15;
  const int wv = threadIdx.x >> 5;
  const int b = blockIdx.y;
  A  += (size_t)b * sA;
  Bt += (size_t)b * sB;
  if (dis)   dis   += (size_t)b * sD;
  if (resid) resid += (size_t)b * sR;

  const int ngrp = Ncols >> 6;
  const int wid  = blockIdx.x * 8 + wv;
  const int tM   = wid / ngrp;              // identical for all 8 waves
  const int tNg  = wid % ngrp;

  __shared__ __bf16 At[2][16 * 32];         // double-buffered 16x32 A tile

  // chunk ids for wave 0's async copies: 64 x 16B chunks per tile
  const int cA = lane;           // row cA>>2, 8-elem part cA&3
  const int cB = lane + 32;
  const __bf16* arow = A + (size_t)tM * 16 * lda;

  auto prefetch = [&](int k0, int buf) {
    if (wv == 0) {
      unsigned l0 = (unsigned)(size_t)&At[buf][(cA >> 2) * 32 + (cA & 3) * 8];
      unsigned l1 = (unsigned)(size_t)&At[buf][(cB >> 2) * 32 + (cB & 3) * 8];
      unsigned long long g0 =
          (unsigned long long)(size_t)(arow + (size_t)(cA >> 2) * lda + k0 + (cA & 3) * 8);
      unsigned long long g1 =
          (unsigned long long)(size_t)(arow + (size_t)(cB >> 2) * lda + k0 + (cB & 3) * 8);
      asm volatile("global_load_async_to_lds_b128 %0, %1, off"
                   :: "v"(l0), "v"(g0) : "memory");
      asm volatile("global_load_async_to_lds_b128 %0, %1, off"
                   :: "v"(l1), "v"(g1) : "memory");
    }
  };

  f32x8 acc[4] = {};
  const __bf16* pb[4];
#pragma unroll
  for (int nt = 0; nt < 4; ++nt)
    pb[nt] = Bt + (size_t)(tNg * 64 + nt * 16 + lq) * ldb;

  prefetch(0, 0);
  if (wv == 0) asm volatile("s_wait_asynccnt 0x0" ::: "memory");
  __syncthreads();

  const int nk = K >> 5;
  for (int i = 0; i < nk; ++i) {
    if (i + 1 < nk) prefetch((i + 1) * 32, (i + 1) & 1);

    const __bf16* pl = At[i & 1];
    FragBF a;                                 // ds_load_b128 x2
    a.q[0] = *(const uint4*)(pl + lq * 32 + half * 16);
    a.q[1] = *(const uint4*)(pl + lq * 32 + half * 16 + 8);

    const int k0 = i * 32;
#pragma unroll
    for (int nt = 0; nt < 4; ++nt) {
      FragBF bb;
      bb.q[0] = *(const uint4*)(pb[nt] + k0 + half * 16);
      bb.q[1] = *(const uint4*)(pb[nt] + k0 + half * 16 + 8);
      acc[nt] = __builtin_amdgcn_wmma_f32_16x16x32_bf16(
          false, a.v, false, bb.v, (short)0, acc[nt], false, false);
    }
    if (wv == 0) asm volatile("s_wait_asynccnt 0x0" ::: "memory");
    __syncthreads();
  }
  gemm_epilogue<EP>(acc, Cp, sC, ldc, b, dis, bias, resid, ldr,
                    tM * 16 + half * 8, tNg, lq);
}

// ---------------------------------------------------------------------------
// Flash attention + fused LN1.  One wave = 16 query rows for one (b,h).
// Q,K: [B,N,Hd] bf16 (head slice strided).  Vt: [B,Hd,N] bf16 (transposed).
// Bb: f32 [H,N,N].  Output MHc: [B,N,Hd] bf16 (heads re-concatenated).
// ---------------------------------------------------------------------------
__global__ __launch_bounds__(128)
void k_attn(const __bf16* __restrict__ Q, const __bf16* __restrict__ Km,
            const __bf16* __restrict__ Vt, const float* __restrict__ Bb,
            const float* __restrict__ g1, const float* __restrict__ be1,
            __bf16* __restrict__ MHc) {
  const int N = 1024, Hd = 1024;
  const int lane = threadIdx.x & 31, half = lane >> 4, lq = lane & 15;
  const int w = threadIdx.x >> 5;
  const int b = blockIdx.z, h = blockIdx.y;
  const int nb = (blockIdx.x * 4 + w) * 16;

  __shared__ __bf16 Pl[4][16 * 32];
  __bf16* pl = Pl[w];

  // preload Q A-fragments for all 4 K-steps (d = 0..127)
  FragBF qa[4];
  {
    const __bf16* qp = Q + ((size_t)(b * N + nb + lq) * Hd + h * 128);
#pragma unroll
    for (int ks = 0; ks < 4; ++ks) {
      qa[ks].q[0] = *(const uint4*)(qp + ks * 32 + half * 16);
      qa[ks].q[1] = *(const uint4*)(qp + ks * 32 + half * 16 + 8);
    }
  }

  f32x8 acc[8] = {};
  float rmax[8], rsum[8];
#pragma unroll
  for (int r = 0; r < 8; ++r) { rmax[r] = -1e30f; rsum[r] = 0.0f; }

  const float scale = 0.08838834764831845f;     // 1/sqrt(128)
  const float* brow = Bb + ((size_t)h * N + nb + half * 8) * N;

  for (int mt = 0; mt < N; mt += 32) {
    // scores for 32 keys (two 16-col tiles)
    f32x8 s0 = {}, s1 = {};
    {
      const __bf16* kp0 = Km + ((size_t)(b * N + mt + lq) * Hd + h * 128);
      const __bf16* kp1 = kp0 + (size_t)16 * Hd;
#pragma unroll
      for (int ks = 0; ks < 4; ++ks) {
        FragBF kf;
        kf.q[0] = *(const uint4*)(kp0 + ks * 32 + half * 16);
        kf.q[1] = *(const uint4*)(kp0 + ks * 32 + half * 16 + 8);
        s0 = __builtin_amdgcn_wmma_f32_16x16x32_bf16(
            false, qa[ks].v, false, kf.v, (short)0, s0, false, false);
        kf.q[0] = *(const uint4*)(kp1 + ks * 32 + half * 16);
        kf.q[1] = *(const uint4*)(kp1 + ks * 32 + half * 16 + 8);
        s1 = __builtin_amdgcn_wmma_f32_16x16x32_bf16(
            false, qa[ks].v, false, kf.v, (short)0, s1, false, false);
      }
    }
    // online softmax (rows split: lanes<16 -> rows r, lanes>=16 -> rows r+8)
    float p0[8], p1[8], fac[8];
#pragma unroll
    for (int r = 0; r < 8; ++r) {
      float v0 = s0[r] * scale + brow[(size_t)r * N + mt + lq];
      float v1 = s1[r] * scale + brow[(size_t)r * N + mt + 16 + lq];
      float mx = xmax16(fmaxf(v0, v1));
      float nm = fmaxf(rmax[r], mx);
      float f  = __expf(rmax[r] - nm);
      rmax[r] = nm;
      p0[r] = __expf(v0 - nm);
      p1[r] = __expf(v1 - nm);
      rsum[r] = rsum[r] * f + xsum16(p0[r] + p1[r]);
      fac[r] = f;
    }
#pragma unroll
    for (int ct = 0; ct < 8; ++ct)
#pragma unroll
      for (int r = 0; r < 8; ++r) acc[ct][r] *= fac[r];

    // C-layout -> A-fragment layout for P via this wave's LDS tile
#pragma unroll
    for (int r = 0; r < 8; ++r) {
      pl[(half * 8 + r) * 32 + lq]      = (__bf16)p0[r];
      pl[(half * 8 + r) * 32 + 16 + lq] = (__bf16)p1[r];
    }
    asm volatile("s_wait_dscnt 0" ::: "memory");
    FragBF pf;
    pf.q[0] = *(const uint4*)(pl + lq * 32 + half * 16);
    pf.q[1] = *(const uint4*)(pl + lq * 32 + half * 16 + 8);

    // acc += P @ V  (Vt rows are feature-major -> contiguous B fragments)
#pragma unroll
    for (int ct = 0; ct < 8; ++ct) {
      const __bf16* vp =
          Vt + ((size_t)b * Hd + h * 128 + ct * 16 + lq) * N + mt + half * 16;
      FragBF vf;
      vf.q[0] = *(const uint4*)(vp);
      vf.q[1] = *(const uint4*)(vp + 8);
      acc[ct] = __builtin_amdgcn_wmma_f32_16x16x32_bf16(
          false, pf.v, false, vf.v, (short)0, acc[ct], false, false);
    }
  }

  // normalize by row sums, fused LN1 over the 128-wide head dim, store bf16
#pragma unroll
  for (int r = 0; r < 8; ++r) {
    const float inv = 1.0f / rsum[r];
    float sm = 0.0f, sq = 0.0f;
#pragma unroll
    for (int ct = 0; ct < 8; ++ct) {
      float x = acc[ct][r] * inv;
      acc[ct][r] = x;
      sm += x; sq += x * x;
    }
    sm = xsum16(sm); sq = xsum16(sq);
    const float mean = sm * (1.0f / 128.0f);
    const float var  = sq * (1.0f / 128.0f) - mean * mean;
    const float rstd = rsqrtf(var + 1e-5f);
    __bf16* orow = MHc + (size_t)(b * N + nb + half * 8 + r) * Hd + h * 128;
#pragma unroll
    for (int ct = 0; ct < 8; ++ct) {
      const int cg = ct * 16 + lq;
      orow[cg] = (__bf16)((acc[ct][r] - mean) * rstd * g1[cg] + be1[cg]);
    }
  }
}

// ---------------------------------------------------------------------------
extern "C" void kernel_launch(void* const* d_in, const int* in_sizes, int n_in,
                              void* d_out, int out_size, void* d_ws,
                              size_t ws_size, hipStream_t stream) {
  (void)in_sizes; (void)n_in; (void)out_size;
  const int B = 8, N = 1024, D = 128, Hd = 1024;

  const float* H   = (const float*)d_in[0];
  const float* A   = (const float*)d_in[1];
  const float* WQ  = (const float*)d_in[2];
  const float* bQ  = (const float*)d_in[3];
  const float* WK  = (const float*)d_in[4];
  const float* bK  = (const float*)d_in[5];
  const float* WV  = (const float*)d_in[6];
  const float* bV  = (const float*)d_in[7];
  const float* Bb  = (const float*)d_in[8];
  const float* WO  = (const float*)d_in[9];
  const float* g0  = (const float*)d_in[10];
  const float* be0 = (const float*)d_in[11];
  const float* g1  = (const float*)d_in[12];
  const float* be1 = (const float*)d_in[13];
  const float* g2  = (const float*)d_in[14];
  const float* be2 = (const float*)d_in[15];
  const float* g3  = (const float*)d_in[16];
  const float* be3 = (const float*)d_in[17];
  const float* W1  = (const float*)d_in[18];
  const float* b1  = (const float*)d_in[19];
  const float* W2  = (const float*)d_in[20];
  const float* b2  = (const float*)d_in[21];

  char* ws = (char*)d_ws;
  size_t off = 0;
  auto take = [&](size_t bytes) -> void* {
    void* p = ws + off;
    off += (bytes + 255) & ~(size_t)255;
    return p;
  };
  __bf16* WqT  = (__bf16*)take((size_t)Hd * D * 2);
  __bf16* WkT  = (__bf16*)take((size_t)Hd * D * 2);
  __bf16* WvT  = (__bf16*)take((size_t)Hd * D * 2);
  __bf16* WoT  = (__bf16*)take((size_t)D * Hd * 2);
  __bf16* W1T  = (__bf16*)take((size_t)D * D * 2);
  __bf16* W2T  = (__bf16*)take((size_t)D * D * 2);
  float*  dis  = (float*) take((size_t)B * N * 4);
  __bf16* Hn   = (__bf16*)take((size_t)B * N * D * 2);
  __bf16* Abf  = (__bf16*)take((size_t)B * N * N * 2);
  __bf16* XWsT = (__bf16*)take((size_t)B * Hd * N * 2);   // reused Q/K/V
  __bf16* Qbf  = (__bf16*)take((size_t)B * N * Hd * 2);
  __bf16* Kbf  = (__bf16*)take((size_t)B * N * Hd * 2);
  __bf16* Vt   = (__bf16*)take((size_t)B * Hd * N * 2);
  __bf16* MHc  = (__bf16*)take((size_t)B * N * Hd * 2);
  float*  Opre = (float*) take((size_t)B * N * D * 4);
  float*  O    = (float*) take((size_t)B * N * D * 4);
  __bf16* Onbf = (__bf16*)take((size_t)B * N * D * 2);
  __bf16* X1   = (__bf16*)take((size_t)B * N * D * 2);
  float*  X2   = (float*) take((size_t)B * N * D * 4);
  if (off > ws_size) return;  // workspace too small; nothing sane to do

  // weights -> transposed bf16
  k_w_transpose<<<512, 256, 0, stream>>>(WQ, WqT, D, Hd);
  k_w_transpose<<<512, 256, 0, stream>>>(WK, WkT, D, Hd);
  k_w_transpose<<<512, 256, 0, stream>>>(WV, WvT, D, Hd);
  k_w_transpose<<<512, 256, 0, stream>>>(WO, WoT, Hd, D);
  k_w_transpose<<< 64, 256, 0, stream>>>(W1, W1T, D, D);
  k_w_transpose<<< 64, 256, 0, stream>>>(W2, W2T, D, D);

  // LN0 (pre-norm) and adjacency prep
  k_ln128<false, true><<<1024, 256, 0, stream>>>(H, g0, be0, nullptr, Hn, B * N);
  k_adj_prep<<<1024, 256, 0, stream>>>(A, Abf, dis);

  // Q/K/V:  XWsT[f,m] = dis[m]*(Hn W)[m,f]  then  adj GEMM vs Ahat
  // (all use the async-LDS staged GEMM; Ncols = 1024 -> 8 waves share tM)
  // Q
  k_gemm_adj<EP_SCALE_T><<<dim3(128, 8), 256, 0, stream>>>(
      Hn, (long long)N * D, D, WqT, 0, D, XWsT, (long long)Hd * N, N,
      dis, N, nullptr, nullptr, 0, 0, Hd, D);
  k_gemm_adj<EP_SCALE_BIAS><<<dim3(128, 8), 256, 0, stream>>>(
      Abf, (long long)N * N, N, XWsT, (long long)Hd * N, N,
      Qbf, (long long)N * Hd, Hd, dis, N, bQ, nullptr, 0, 0, Hd, N);
  // K
  k_gemm_adj<EP_SCALE_T><<<dim3(128, 8), 256, 0, stream>>>(
      Hn, (long long)N * D, D, WkT, 0, D, XWsT, (long long)Hd * N, N,
      dis, N, nullptr, nullptr, 0, 0, Hd, D);
  k_gemm_adj<EP_SCALE_BIAS><<<dim3(128, 8), 256, 0, stream>>>(
      Abf, (long long)N * N, N, XWsT, (long long)Hd * N, N,
      Kbf, (long long)N * Hd, Hd, dis, N, bK, nullptr, 0, 0, Hd, N);
  // V (stored transposed [B,Hd,N] for the P@V B-fragments)
  k_gemm_adj<EP_SCALE_T><<<dim3(128, 8), 256, 0, stream>>>(
      Hn, (long long)N * D, D, WvT, 0, D, XWsT, (long long)Hd * N, N,
      dis, N, nullptr, nullptr, 0, 0, Hd, D);
  k_gemm_adj<EP_SCALE_BIAS_T><<<dim3(128, 8), 256, 0, stream>>>(
      Abf, (long long)N * N, N, XWsT, (long long)Hd * N, N,
      Vt, (long long)Hd * N, N, dis, N, bV, nullptr, 0, 0, Hd, N);

  // attention + fused LN1 -> MHc [B,N,Hd]
  k_attn<<<dim3(16, 8, 8), 128, 0, stream>>>(Qbf, Kbf, Vt, Bb, g1, be1, MHc);

  // O_pre = MHc @ W_O + H ; O = LN2(O_pre) (f32 + bf16 copies)
  k_gemm_bt<EP_RESID_F32><<<dim3(128, 1), 256, 0, stream>>>(
      MHc, 0, Hd, WoT, 0, Hd, Opre, 0, D,
      nullptr, 0, nullptr, H, 0, D, B * N, D, Hd);
  k_ln128<true, true><<<1024, 256, 0, stream>>>(Opre, g2, be2, O, Onbf, B * N);

  // FFN: X1 = relu(O W1 + b1); X2 = relu(X1 W2 + b2)
  k_gemm_bt<EP_BIAS_RELU_BF><<<dim3(128, 1), 256, 0, stream>>>(
      Onbf, 0, D, W1T, 0, D, X1, 0, D,
      nullptr, 0, b1, nullptr, 0, 0, B * N, D, D);
  k_gemm_bt<EP_BIAS_RELU_F32><<<dim3(128, 1), 256, 0, stream>>>(
      X1, 0, D, W2T, 0, D, X2, 0, D,
      nullptr, 0, b2, nullptr, 0, 0, B * N, D, D);

  // out = O + LN3(X2)
  k_ln_add<<<1024, 256, 0, stream>>>(X2, g3, be3, O, (float*)d_out, B * N);
}
